// MyModel_87522843559291
// MI455X (gfx1250) — matrix operand order repaired and checked by reading, not compiled, not run on hardware
//
#include <hip/hip_runtime.h>
#include <hip/hip_bf16.h>

typedef __attribute__((ext_vector_type(16))) _Float16 v16h;
typedef __attribute__((ext_vector_type(8)))  _Float16 v8h;
typedef __attribute__((ext_vector_type(8)))  float    v8f;

#define BATCH 1024
#define SEQ   64
#define DIM   512
#define VOCAB 300

// ---------------- fragment helpers (wave32 WMMA layouts, gfx1250) ----------------

// A-matrix 16x32 f16, source row-major [row][k] with leading dim ld (halfs).
// lanes 0-15: row = row0+lane, K = k0+0..7 (v0..3) and k0+16..23 (v4..7)
// lanes 16-31: same rows, K = k0+8..15 and k0+24..31
__device__ inline v16h load_frag_a(const _Float16* base, int ld, int row0, int k0, int lane) {
    int r = row0 + (lane & 15);
    int k = k0 + ((lane >> 4) << 3);
    union { v16h v; v8h h[2]; } u;
    u.h[0] = *(const v8h*)(base + r * ld + k);
    u.h[1] = *(const v8h*)(base + r * ld + k + 16);
    return u.v;
}

// B-matrix 32x16 f16, source stored as [N][K] (column of B contiguous in K), ld in halfs.
// lanes 0-15: col = n0+lane, K = k0+0..7 and k0+16..23 ; lanes 16-31: same col, K = k0+8..15 and k0+24..31
__device__ inline v16h load_frag_b(const _Float16* base, int ld, int n0, int k0, int lane) {
    int n = n0 + (lane & 15);
    int k = k0 + ((lane >> 4) << 3);
    union { v16h v; v8h h[2]; } u;
    u.h[0] = *(const v8h*)(base + n * ld + k);
    u.h[1] = *(const v8h*)(base + n * ld + k + 16);
    return u.v;
}

__device__ inline v8f wmma_f16(v16h a, v16h b, v8f c) {
    return __builtin_amdgcn_wmma_f32_16x16x32_f16(false, a, false, b, (short)0, c, false, false);
}

// C/D layout: lane n (0-15): VGPR i = (M=mbase+i, N=nbase+n); lane n+16: M=mbase+8+i.
__device__ inline void store_qkv(const v8f& c, _Float16* qkv, _Float16* qkvT,
                                 int mbase, int nbase, int lane) {
    int col  = nbase + (lane & 15);
    int row0 = mbase + ((lane >> 4) << 3);
    v8h hv;
#pragma unroll
    for (int i = 0; i < 8; ++i) hv[i] = (_Float16)c[i];
    *(v8h*)(qkvT + col * SEQ + row0) = hv;          // one ds_store_b128
#pragma unroll
    for (int i = 0; i < 8; ++i) qkv[(row0 + i) * DIM + col] = hv[i];
}

// ---------------- precompute kernels ----------------

__global__ void cvt_emb_kernel(const float* __restrict__ src, _Float16* __restrict__ dst, int n) {
    int i = blockIdx.x * blockDim.x + threadIdx.x;
    if (i < n) dst[i] = (_Float16)src[i];
}

__global__ void transpose_q_kernel(const float* __restrict__ q, _Float16* __restrict__ qt) {
    int i = blockIdx.x * blockDim.x + threadIdx.x;   // 512*512 elements
    int k = i >> 9;
    int n = i & 511;
    qt[n * DIM + k] = (_Float16)q[k * DIM + n];
}

// ---------------- main fused attention kernel ----------------
// 1 block = 1 batch; 256 threads = 8 wave32.
// LDS: X[64][512]h | qkv[64][512]h | qkvT[512][64]h | scores[64][64]h | mask[64]f | ids_en[64]i

#define LDS_X      0
#define LDS_QKV    65536
#define LDS_QKVT   131072
#define LDS_SCORES 196608
#define LDS_MASK   204800
#define LDS_IDSEN  205056
#define LDS_TOTAL  205312

__global__ __launch_bounds__(256)
void attn_kernel(const int* __restrict__ ids_en_g, const int* __restrict__ ids_fr_g,
                 const float* __restrict__ emb_f32, const _Float16* __restrict__ emb_h,
                 const _Float16* __restrict__ qt_h, float* __restrict__ out) {
    extern __shared__ __align__(16) unsigned char smem[];
    _Float16* X     = (_Float16*)(smem + LDS_X);
    _Float16* qkv   = (_Float16*)(smem + LDS_QKV);
    _Float16* qkvT  = (_Float16*)(smem + LDS_QKVT);
    _Float16* scr   = (_Float16*)(smem + LDS_SCORES);
    float*    maskv = (float*)(smem + LDS_MASK);
    int*      idsen = (int*)(smem + LDS_IDSEN);

    const int b    = blockIdx.x;
    const int tid  = threadIdx.x;
    const int lane = tid & 31;
    const int wave = tid >> 5;

    // ---- Phase 1: mask, ids, gather embedded_fr -> X via async global->LDS DMA ----
    if (tid < SEQ) {
        int idf = ids_fr_g[b * SEQ + tid];
        maskv[tid] = (idf != 0) ? 1.0f : 0.0f;
        idsen[tid] = ids_en_g[b * SEQ + tid];
    }
    {
        int row = tid >> 2;               // 4 threads per row
        int cb  = (tid & 3) * 128;        // 128 halfs each
        int idf = ids_fr_g[b * SEQ + row];
        // LDS byte address = low 32 bits of the generic shared pointer
        unsigned lds_a = (unsigned)(uintptr_t)(void*)(X + row * DIM + cb);
        unsigned long long ga = (unsigned long long)(uintptr_t)(emb_h + (size_t)idf * DIM + cb);
#pragma unroll
        for (int i = 0; i < 16; ++i) {
            asm volatile("global_load_async_to_lds_b128 %0, %1, off"
                         :: "v"(lds_a), "v"(ga) : "memory");
            lds_a += 16;
            ga    += 16;
        }
        asm volatile("s_wait_asynccnt 0x0" ::: "memory");
    }
    __syncthreads();

    // ---- Phase 2: qkv = X @ Qdense  (M=64, N=512, K=512) ----
    // B fragments stream from global Qt (L2-resident, 512 KB), software-pipelined one
    // kc ahead so the L2 round-trip overlaps the 4 WMMAs + 8 ds_load_b128 of the
    // current step.
#pragma unroll 1
    for (int nt = 0; nt < 4; ++nt) {
        int nbase = (wave * 4 + nt) * 16;
        v8f acc[4] = {};
        v16h bf = load_frag_b(qt_h, DIM, nbase, 0, lane);
#pragma unroll 1
        for (int kc = 0; kc < 15; ++kc) {
            int k0 = kc * 32;
            v16h bfn = load_frag_b(qt_h, DIM, nbase, k0 + 32, lane);   // prefetch next
#pragma unroll
            for (int m = 0; m < 4; ++m) {
                v16h af = load_frag_a(X, DIM, m * 16, k0, lane);
                acc[m] = wmma_f16(af, bf, acc[m]);
            }
            bf = bfn;
        }
#pragma unroll
        for (int m = 0; m < 4; ++m) {                                  // epilogue kc = 15
            v16h af = load_frag_a(X, DIM, m * 16, 480, lane);
            acc[m] = wmma_f16(af, bf, acc[m]);
        }
#pragma unroll
        for (int m = 0; m < 4; ++m) store_qkv(acc[m], qkv, qkvT, m * 16, nbase, lane);
    }
    __syncthreads();

    // ---- Phase 3: scores = (qkv @ qkvT) * mask  (M=64, N=64, K=512) ----
#pragma unroll 1
    for (int tt = 0; tt < 2; ++tt) {
        int t     = wave * 2 + tt;        // 16 tiles over 8 waves
        int mbase = (t >> 2) * 16;
        int nbase = (t & 3) * 16;
        v8f c = {};
#pragma unroll 1
        for (int kc = 0; kc < 16; ++kc) {
            int k0 = kc * 32;
            v16h af = load_frag_a(qkv, DIM, mbase, k0, lane);
            v16h bf = load_frag_b(qkv, DIM, nbase, k0, lane);  // row k of qkv = column k of K^T
            c = wmma_f16(af, bf, c);
        }
        int col  = nbase + (lane & 15);
        int row0 = mbase + ((lane >> 4) << 3);
        float mk = maskv[col];
#pragma unroll
        for (int i = 0; i < 8; ++i) scr[(row0 + i) * SEQ + col] = (_Float16)(c[i] * mk);
    }
    __syncthreads();

    // ---- Phase 4: encoded = scores @ qkv (M=64, N=512, K=64); out = emb[ids_en] + encoded ----
#pragma unroll 1
    for (int nt = 0; nt < 4; ++nt) {
        int nbase = (wave * 4 + nt) * 16;
        v8f acc[4] = {};
#pragma unroll
        for (int kc = 0; kc < 2; ++kc) {
            int k0 = kc * 32;
            v16h bf = load_frag_b(qkvT, SEQ, nbase, k0, lane);  // qkvT[n][k], K-contiguous
#pragma unroll
            for (int m = 0; m < 4; ++m) {
                v16h af = load_frag_a(scr, SEQ, m * 16, k0, lane);
                acc[m] = wmma_f16(af, bf, acc[m]);
            }
        }
        int col = nbase + (lane & 15);
#pragma unroll
        for (int m = 0; m < 4; ++m) {
            int row0 = m * 16 + ((lane >> 4) << 3);
#pragma unroll
            for (int i = 0; i < 8; ++i) {
                int r = row0 + i;
                size_t oidx = ((size_t)b * SEQ + r) * DIM + col;
                out[oidx] = emb_f32[(size_t)idsen[r] * DIM + col] + acc[m][i];
            }
        }
    }
}

// ---------------- launch ----------------

extern "C" void kernel_launch(void* const* d_in, const int* in_sizes, int n_in,
                              void* d_out, int out_size, void* d_ws, size_t ws_size,
                              hipStream_t stream) {
    const int*   ids_en = (const int*)d_in[0];    // [1024, 64]
    const int*   ids_fr = (const int*)d_in[1];    // [1024, 64]
    const float* emb    = (const float*)d_in[2];  // [300, 512]
    const float* qd     = (const float*)d_in[3];  // [512, 512]
    float*       out    = (float*)d_out;          // [1024, 64, 512]

    _Float16* emb_h = (_Float16*)d_ws;                                  // 300*512*2 = 307200 B
    _Float16* qt_h  = (_Float16*)((unsigned char*)d_ws + 307200);       // 512*512*2 = 524288 B

    cvt_emb_kernel<<<(VOCAB * DIM + 255) / 256, 256, 0, stream>>>(emb, emb_h, VOCAB * DIM);
    transpose_q_kernel<<<(DIM * DIM) / 256, 256, 0, stream>>>(qd, qt_h);
    attn_kernel<<<BATCH, 256, LDS_TOTAL, stream>>>(ids_en, ids_fr, emb, emb_h, qt_h, out);
}
